// ConstraintMessagePassing_15642270892858
// MI455X (gfx1250) — matrix operand order repaired and checked by reading, not compiled, not run on hardware
//
#include <hip/hip_runtime.h>

// CDNA5 / gfx1250 fused constraint-message-passing kernel (round 2).
// wave32, v_wmma_f32_16x16x32_f16 for all GEMMs.
// Feature axes of the layer-1 weight matrices are permuted so that A-matrix
// fragments can be assembled directly in registers (no LDS staging panel).

typedef __attribute__((ext_vector_type(16))) _Float16 v16h;
typedef __attribute__((ext_vector_type(8)))  _Float16 v8h;
typedef __attribute__((ext_vector_type(8)))  float    v8f;

#define BN   64
#define HH   128
#define WW   128
#define CC   10
#define FDIM 32

static __device__ const int kDY[8] = {-1,-1,-1, 0, 0, 1, 1, 1};
static __device__ const int kDX[8] = {-1, 0, 1,-1, 1,-1, 0, 1};

static __device__ __forceinline__ v16h cat16(v8h lo, v8h hi) {
  return __builtin_shufflevector(lo, hi, 0,1,2,3,4,5,6,7,8,9,10,11,12,13,14,15);
}
// A-fragment loader from a row-major f16 row: lane l holds row (l&15);
// elements 0..7 = K kb..kb+7, elements 8..15 = K kb+16..kb+23, kb=(l>>4)*8.
static __device__ __forceinline__ v16h afrag(const _Float16* row, int kb) {
  v8h c0 = *(const v8h*)(row + kb);
  v8h c1 = *(const v8h*)(row + kb + 16);
  return cat16(c0, c1);
}
// B-fragment: lane l holds column n; K kb..kb+7 and kb+16..kb+23, kb=(l>>4)*8.
static __device__ __forceinline__ v16h bfrag(const _Float16* s) {
  v8h c0 = *(const v8h*)s;
  v8h c1 = *(const v8h*)(s + 16);
  return cat16(c0, c1);
}
static __device__ __forceinline__ v8f wmma_f16(v16h a, v16h b, v8f c) {
  return __builtin_amdgcn_wmma_f32_16x16x32_f16(false, a, false, b, (short)0, c, false, false);
}

__global__ void __launch_bounds__(256)
cmp_fused_kernel(const float* __restrict__ beliefs,
                 const float* __restrict__ cns,
                 const float* __restrict__ fnv,
                 const float* __restrict__ mw1, const float* __restrict__ mb1,
                 const float* __restrict__ mw2, const float* __restrict__ mb2,
                 const float* __restrict__ uw1, const float* __restrict__ ub1,
                 const float* __restrict__ uw2, const float* __restrict__ ub2,
                 const float* __restrict__ gma, const float* __restrict__ bta,
                 float* __restrict__ out)
{
  // ---- LDS (~52 KB) ----
  __shared__ __align__(16) _Float16 sW1T[64][64];  // msg L1, N-major, K permuted (see fill)
  __shared__ __align__(16) _Float16 sW2T[16][64];  // msg L2, Npad 10->16
  __shared__ __align__(16) _Float16 sU1T[64][32];  // upd L1, K permuted
  __shared__ __align__(16) _Float16 sU2T[16][64];  // upd L2
  __shared__ float sB1[64], sUB1[64];
  __shared__ float sB2[16], sUB2[16], sG[16], sBt[16];
  __shared__ float sBel[18*18*10];                 // replicate-padded halo (fp32)
  __shared__ __align__(16) _Float16 sFn[FDIM];
  __shared__ __align__(16) _Float16 sAgg[256*10];  // aggregated messages per pixel
  __shared__ __align__(16) unsigned char sScrRaw[8][2048]; // per-wave H / D scratch

  const int tid  = threadIdx.x;
  const int lane = tid & 31;
  const int wv   = tid >> 5;
  const int tx = blockIdx.x, ty = blockIdx.y, bimg = blockIdx.z;

  __builtin_prefetch(beliefs + (((size_t)bimg*HH + ty*16)*WW + tx*16)*CC, 0, 1);

  // ================= Phase 1: stage halo + weights =================
  for (int idx = tid; idx < 18*18*10; idx += 256) {
    int hp = idx / 10, ch = idx - hp*10;
    int hy = hp / 18, hx = hp - hy*18;
    int gy = ty*16 + hy - 1; gy = gy < 0 ? 0 : (gy > HH-1 ? HH-1 : gy);
    int gx = tx*16 + hx - 1; gx = gx < 0 ? 0 : (gx > WW-1 ? WW-1 : gx);
    sBel[idx] = beliefs[(((size_t)bimg*HH + gy)*WW + gx)*CC + ch];
  }
  if (tid < FDIM) sFn[tid] = (_Float16)fnv[bimg*FDIM + tid];

  // msg L1 weights with K-permutation:
  //  k 0..7  -> self0..7      k 8..15 -> neigh0..7
  //  k 16,17 -> self8,self9   k 18..20 -> cons0..2
  //  k 24,25 -> neigh8,neigh9 k 32..63 -> fn0..31   (rest = 0 pad)
  for (int idx = tid; idx < 64*64; idx += 256) {
    int n = idx >> 6, k = idx & 63;
    int f;
    if (k < 8)            f = k;
    else if (k < 16)      f = 10 + (k - 8);
    else if (k == 16)     f = 8;
    else if (k == 17)     f = 9;
    else if (k <= 20)     f = 20 + (k - 18);
    else if (k == 24)     f = 18;
    else if (k == 25)     f = 19;
    else if (k >= 32)     f = 23 + (k - 32);
    else                  f = -1;
    sW1T[n][k] = (f >= 0) ? (_Float16)mw1[f*64 + n] : (_Float16)0.f;
  }
  for (int idx = tid; idx < 16*64; idx += 256) {
    int n = idx >> 6, k = idx & 63;
    sW2T[n][k] = (n < 10) ? (_Float16)mw2[k*10 + n] : (_Float16)0.f;
  }
  // upd L1 weights, K-permutation:
  //  k 0..7 -> bel0..7   k 8..15 -> agg0..7
  //  k 16,17 -> bel8,bel9  k 18,19 -> agg8,agg9  (rest = 0 pad)
  for (int idx = tid; idx < 64*32; idx += 256) {
    int n = idx >> 5, k = idx & 31;
    int f;
    if (k < 8)        f = k;
    else if (k < 16)  f = 10 + (k - 8);
    else if (k == 16) f = 8;
    else if (k == 17) f = 9;
    else if (k == 18) f = 18;
    else if (k == 19) f = 19;
    else              f = -1;
    sU1T[n][k] = (f >= 0) ? (_Float16)uw1[f*64 + n] : (_Float16)0.f;
  }
  for (int idx = tid; idx < 16*64; idx += 256) {
    int n = idx >> 6, k = idx & 63;
    sU2T[n][k] = (n < 10) ? (_Float16)uw2[k*10 + n] : (_Float16)0.f;
  }
  if (tid < 64) { sB1[tid] = mb1[tid]; sUB1[tid] = ub1[tid]; }
  if (tid < 16) {
    sB2[tid]  = tid < 10 ? mb2[tid] : 0.f;
    sUB2[tid] = tid < 10 ? ub2[tid] : 0.f;
    sG[tid]   = tid < 10 ? gma[tid] : 1.f;
    sBt[tid]  = tid < 10 ? bta[tid] : 0.f;
  }
  __syncthreads();

  // ================= Phase 2: message net =================
  _Float16* sA = (_Float16*)&sScrRaw[wv][0];     // 16x64 f16 H panel
  const int rB    = lane & 15;
  const int kb    = (lane >> 4) * 8;
  const int bko   = (lane >> 4) * 8;
  const int rbase = (lane >> 4) * 8;
  const int g8    = (lane >> 4) * 8;

  // hoisted loop-invariant operands
  v16h wB1[4][2], wB2[2];
#pragma unroll
  for (int nt = 0; nt < 4; ++nt) {
    wB1[nt][0] = bfrag(&sW1T[nt*16 + rB][bko]);
    wB1[nt][1] = bfrag(&sW1T[nt*16 + rB][32 + bko]);
  }
  wB2[0] = bfrag(&sW2T[rB][bko]);
  wB2[1] = bfrag(&sW2T[rB][32 + bko]);
  // a1 (K=32..63) is pure filename-embed: invariant over the whole phase
  v16h a1 = cat16(*(const v8h*)&sFn[g8], *(const v8h*)&sFn[16 + g8]);
  float b1r[4];
#pragma unroll
  for (int nt = 0; nt < 4; ++nt) b1r[nt] = sB1[nt*16 + rB];
  const float b2r = sB2[rB];

  for (int t = wv*16; t < wv*16 + 16; ++t) {     // 128 row-tiles / 8 waves
    // --- A-frag (K=0..31) built directly in registers ---
    int R = t*16 + rB;
    int p = R >> 3, j = R & 7;
    int py = p >> 4, px = p & 15;
    const float* bsrc = &sBel[((lane < 16) ? ((py+1)*18 + (px+1))
                                           : ((py+1+kDY[j])*18 + (px+1+kDX[j])))*10];
    float c0v = 0.f, c1v = 0.f, c2v = 0.f;
    if (lane < 16) {  // constraints consumed only by the low lane-half
      const float* cp = cns + (((size_t)bimg*HH + (ty*16+py))*WW + (tx*16+px))*24 + j*3;
      c0v = cp[0]; c1v = cp[1]; c2v = cp[2];
    }
    v16h a0;
#pragma unroll
    for (int i = 0; i < 8; ++i) a0[i] = (_Float16)bsrc[i];
    a0[8]  = (_Float16)bsrc[8];
    a0[9]  = (_Float16)bsrc[9];
    a0[10] = (_Float16)c0v;
    a0[11] = (_Float16)c1v;
    a0[12] = (_Float16)c2v;
    a0[13] = (_Float16)0.f;
    a0[14] = (_Float16)0.f;
    a0[15] = (_Float16)0.f;

    // --- layer 1: 8 WMMAs ---
    v8f h[4];
#pragma unroll
    for (int nt = 0; nt < 4; ++nt) {
      v8f c = {};
      c = wmma_f16(a0, wB1[nt][0], c);
      c = wmma_f16(a1, wB1[nt][1], c);
      float bias = b1r[nt];
#pragma unroll
      for (int i = 0; i < 8; ++i) { float v2 = c[i] + bias; c[i] = v2 > 0.f ? v2 : 0.f; }
      h[nt] = c;
    }
    // --- transpose H (D-layout -> A-layout) via per-wave LDS ---
#pragma unroll
    for (int nt = 0; nt < 4; ++nt) {
      int coln = nt*16 + rB;
#pragma unroll
      for (int i = 0; i < 8; ++i) sA[(rbase + i)*64 + coln] = (_Float16)h[nt][i];
    }
    asm volatile("s_wait_dscnt 0x0" ::: "memory");

    // --- layer 2: 2 WMMAs ---
    const _Float16* hrow = &sA[rB*64];
    v16h ha0 = afrag(hrow, kb);
    v16h ha1 = afrag(hrow + 32, kb);
    v8f d = {};
    d = wmma_f16(ha0, wB2[0], d);
    d = wmma_f16(ha1, wB2[1], d);

    // --- neighbor mean: 8 message rows live in this lane's accumulators ---
    float sm  = d[0]+d[1]+d[2]+d[3]+d[4]+d[5]+d[6]+d[7];
    float agg = sm * 0.125f + b2r;
    int pp = 2*t + (lane >> 4);
    if (rB < 10) sAgg[pp*10 + rB] = (_Float16)agg;
  }
  __syncthreads();

  // ================= Phase 3: update net + residual + LayerNorm =================
  float* sD = (float*)&sScrRaw[wv][0];           // 16x16 f32 view (aliases scratch)
  v16h uB1[4], uB2[2];
#pragma unroll
  for (int nt = 0; nt < 4; ++nt) uB1[nt] = bfrag(&sU1T[nt*16 + rB][bko]);
  uB2[0] = bfrag(&sU2T[rB][bko]);
  uB2[1] = bfrag(&sU2T[rB][32 + bko]);
  float ub1r[4];
#pragma unroll
  for (int nt = 0; nt < 4; ++nt) ub1r[nt] = sUB1[nt*16 + rB];
  const float ub2r = sUB2[rB];

  for (int s = 0; s < 2; ++s) {
    int t2 = wv*2 + s;                            // pixels t2*16 .. t2*16+15
    int p  = t2*16 + rB;
    int py = p >> 4, px = p & 15;
    const float*    bel = &sBel[((py+1)*18 + (px+1))*10];
    const _Float16* ag  = &sAgg[p*10];
    v16h ua;
    if (lane < 16) {           // {bel0..7 | bel8,bel9,agg8,agg9,pad}
#pragma unroll
      for (int i = 0; i < 8; ++i) ua[i] = (_Float16)bel[i];
      ua[8]  = (_Float16)bel[8];
      ua[9]  = (_Float16)bel[9];
      ua[10] = ag[8];
      ua[11] = ag[9];
      ua[12] = (_Float16)0.f; ua[13] = (_Float16)0.f;
      ua[14] = (_Float16)0.f; ua[15] = (_Float16)0.f;
    } else {                   // {agg0..7 | pad}
#pragma unroll
      for (int i = 0; i < 8; ++i) ua[i] = ag[i];
#pragma unroll
      for (int i = 8; i < 16; ++i) ua[i] = (_Float16)0.f;
    }

    v8f g[4];
#pragma unroll
    for (int nt = 0; nt < 4; ++nt) {
      v8f c = {};
      c = wmma_f16(ua, uB1[nt], c);
      float bias = ub1r[nt];
#pragma unroll
      for (int i = 0; i < 8; ++i) { float v2 = c[i] + bias; c[i] = v2 > 0.f ? v2 : 0.f; }
      g[nt] = c;
    }
    _Float16* sA2 = (_Float16*)&sScrRaw[wv][0];
#pragma unroll
    for (int nt = 0; nt < 4; ++nt) {
      int coln = nt*16 + rB;
#pragma unroll
      for (int i = 0; i < 8; ++i) sA2[(rbase + i)*64 + coln] = (_Float16)g[nt][i];
    }
    asm volatile("s_wait_dscnt 0x0" ::: "memory");

    const _Float16* hrow = &sA2[rB*64];
    v16h ha0 = afrag(hrow, kb);
    v16h ha1 = afrag(hrow + 32, kb);
    v8f d = {};
    d = wmma_f16(ha0, uB2[0], d);
    d = wmma_f16(ha1, uB2[1], d);
    {
#pragma unroll
      for (int i = 0; i < 8; ++i) sD[(rbase + i)*16 + rB] = d[i] + ub2r;
    }
    asm volatile("s_wait_dscnt 0x0" ::: "memory");

    if (lane < 16) {                              // one pixel per lane: LN + store
      int pl = t2*16 + lane;
      int ply = pl >> 4, plx = pl & 15;
      const float* belr = &sBel[((ply+1)*18 + (plx+1))*10];
      const float* drow = &sD[lane*16];
      float x[10], mu = 0.f;
#pragma unroll
      for (int c = 0; c < 10; ++c) { x[c] = belr[c] + 0.5f*drow[c]; mu += x[c]; }
      mu *= 0.1f;
      float var = 0.f;
#pragma unroll
      for (int c = 0; c < 10; ++c) { float dv = x[c] - mu; var += dv*dv; }
      var *= 0.1f;
      float inv = rsqrtf(var + 1e-5f);
      float* o = out + (((size_t)bimg*HH + (ty*16+ply))*WW + (tx*16+plx))*CC;
#pragma unroll
      for (int c = 0; c < 10; ++c) o[c] = (x[c]-mu)*inv*sG[c] + sBt[c];
    }
  }
}

extern "C" void kernel_launch(void* const* d_in, const int* in_sizes, int n_in,
                              void* d_out, int out_size, void* d_ws, size_t ws_size,
                              hipStream_t stream)
{
  (void)in_sizes; (void)n_in; (void)out_size; (void)d_ws; (void)ws_size;
  const float* beliefs = (const float*)d_in[0];
  const float* cnsP    = (const float*)d_in[1];
  const float* fnv     = (const float*)d_in[2];
  const float* mw1 = (const float*)d_in[3];
  const float* mb1 = (const float*)d_in[4];
  const float* mw2 = (const float*)d_in[5];
  const float* mb2 = (const float*)d_in[6];
  const float* uw1 = (const float*)d_in[7];
  const float* ub1 = (const float*)d_in[8];
  const float* uw2 = (const float*)d_in[9];
  const float* ub2 = (const float*)d_in[10];
  const float* gma = (const float*)d_in[11];
  const float* bta = (const float*)d_in[12];
  float* outp = (float*)d_out;

  dim3 grid(WW/16, HH/16, BN);   // 8 x 8 x 64 tiles
  dim3 block(256);               // 8 waves (wave32)
  hipLaunchKernelGGL(cmp_fused_kernel, grid, block, 0, stream,
                     beliefs, cnsP, fnv, mw1, mb1, mw2, mb2,
                     uw1, ub1, uw2, ub2, gma, bta, outp);
}
